// ODE_RNN_79104707657778
// MI455X (gfx1250) — matrix operand order repaired and checked by reading, not compiled, not run on hardware
//
#include <hip/hip_runtime.h>

typedef __attribute__((ext_vector_type(16))) __bf16 v16bf;
typedef __attribute__((ext_vector_type(8)))  float  v8f;

// ---------------- Tsit5 coefficients ----------------
__constant__ float CTAB[6] = {0.0f, 0.161f, 0.327f, 0.9f, 0.9800255409045097f, 1.0f};
__constant__ float ATAB[6][5] = {
    {0.f, 0.f, 0.f, 0.f, 0.f},
    {0.161f, 0.f, 0.f, 0.f, 0.f},
    {-0.008480655492356989f, 0.335480655492357f, 0.f, 0.f, 0.f},
    {2.8971530571054935f, -6.359448489975075f, 4.3622954328695815f, 0.f, 0.f},
    {5.325864828439257f, -11.748883564062828f, 7.4955393428898365f, -0.09249506636175525f, 0.f},
    {5.86145544294642f, -12.92096931784711f, 8.159367898576159f, -0.071584973281401f, -0.028269050394068383f}};
__constant__ float BTAB[6] = {0.09646076681806523f, 0.01f, 0.4798896504144996f,
                              1.379008574103742f, -3.290069515436081f, 2.324710524099774f};

// ---------------- packed-weight workspace layout (ushort elements) ----------------
// B-operand tiles are 16x32 bf16 = 512 elements, lane-packed.
constexpr int OFF_W0  = 0;                      // ode_w0 : N=256, Kpad=160 -> 16*5 tiles
constexpr int OFF_W1  = OFF_W0  + 16 * 5 * 512; // ode_w1 : N=256, K=256   -> 16*8 tiles
constexpr int OFF_W2  = OFF_W1  + 16 * 8 * 512; // ode_w2 : N=128, K=256   -> 8*8 tiles
constexpr int OFF_RNN = OFF_W2  +  8 * 8 * 512; // [Wh | Wx]: N=128, Kpad=160 -> 8*5 tiles
constexpr int OFF_INJ = OFF_RNN +  8 * 5 * 512; // inj_w : N=128, Kpad=160 -> 8*5 tiles
constexpr int OFF_O0  = OFF_INJ +  8 * 5 * 512; // out_w0: N=256, K=128    -> 16*4 tiles
constexpr int OFF_O1  = OFF_O0  + 16 * 4 * 512; // out_w1: N=256, K=256    -> 16*8 tiles
constexpr int OFF_O2  = OFF_O1  + 16 * 8 * 512; // out_w2: N=16,  K=256    -> 1*8 tiles
// total = 282624 ushorts = 552 KB of d_ws

// ---------------- bf16 helpers (round-to-nearest-even) ----------------
__device__ __forceinline__ unsigned short f2bfu(float x) {
    unsigned u = __float_as_uint(x);
    return (unsigned short)((u + 0x7FFFu + ((u >> 16) & 1u)) >> 16);
}
__device__ __forceinline__ float bf2f(unsigned short s) {
    return __uint_as_float(((unsigned)s) << 16);
}

// ---------------- fast activations on the CDNA5 TRANS pipe ----------------
__device__ __forceinline__ float fast_tanh(float x) {
#if __has_builtin(__builtin_amdgcn_tanhf)
    return __builtin_amdgcn_tanhf(x);      // v_tanh_f32 (gfx1250 hardware transcendental)
#else
    return tanhf(x);
#endif
}
__device__ __forceinline__ float fast_silu(float x) {
#if __has_builtin(__builtin_amdgcn_exp2f) && __has_builtin(__builtin_amdgcn_rcpf)
    // silu(x) = x * sigmoid(x); sigmoid via raw v_exp_f32 (2^t) + v_rcp_f32
    return x * __builtin_amdgcn_rcpf(1.f + __builtin_amdgcn_exp2f(-1.4426950408889634f * x));
#else
    return x / (1.f + __expf(-x));
#endif
}

// A-operand packed write: 16x32 bf16 tile layout per ISA 7.12.2
// lane = m + 16*bit3(kk); vgpr j = 4*bit4(kk) + (kk&7)/2 ; half = kk&1
__device__ __forceinline__ void panel_write(unsigned short* p, int m, int k, unsigned short v) {
    const int kt = k >> 5, kk = k & 31;
    const int lane = m + (((kk >> 3) & 1) << 4);
    const int j = ((kk >> 4) << 2) + ((kk & 7) >> 1);
    p[(kt << 9) + (lane << 4) + (j << 1) + (kk & 1)] = v;
}

// one 16x16 output tile: A panel in LDS (lane-packed), W tiles in global (lane-packed).
// Two independent accumulators (even/odd K-tiles) halve the chained WMMA RAW latency.
__device__ __forceinline__ v8f gemm16(const unsigned short* pan, const unsigned short* wt,
                                      int kTiles, int lane) {
    v8f acc0 = {0.f, 0.f, 0.f, 0.f, 0.f, 0.f, 0.f, 0.f};
    v8f acc1 = {0.f, 0.f, 0.f, 0.f, 0.f, 0.f, 0.f, 0.f};
    #pragma unroll
    for (int kt = 0; kt < kTiles; ++kt) {
        const v16bf a = *reinterpret_cast<const v16bf*>(pan + (kt << 9) + (lane << 4));
        const v16bf b = *reinterpret_cast<const v16bf*>(wt + (kt << 9) + (lane << 4));
        if (kt & 1)
            acc1 = __builtin_amdgcn_wmma_f32_16x16x32_bf16(false, a, false, b, (short)0, acc1,
                                                           false, false);
        else
            acc0 = __builtin_amdgcn_wmma_f32_16x16x32_bf16(false, a, false, b, (short)0, acc0,
                                                           false, false);
    }
    return acc0 + acc1;
}

// ---------------- weight pack: fp32 [N][Ksrc] row-major -> bf16 B-operand tiles ----------------
// B tile (K x 16): lane = nl + 16*bit3(kk); vgpr j = 4*bit4(kk) + (kk&7)/2; half = kk&1
__global__ void pack_weight(const float* __restrict__ src, unsigned short* __restrict__ dst,
                            int N, int Ksrc, int kOff, int kTilesTotal, int Kspan) {
    const int idx = blockIdx.x * blockDim.x + threadIdx.x;
    if (idx >= N * Kspan) return;
    const int n = idx / Kspan, kl = idx - n * Kspan;
    const int kG = kOff + kl;
    const float v = (kl < Ksrc) ? src[n * Ksrc + kl] : 0.f;
    const int nt = n >> 4, nl = n & 15, kt = kG >> 5, kk = kG & 31;
    const int lane = nl + (((kk >> 3) & 1) << 4);
    const int j = ((kk >> 4) << 2) + ((kk & 7) >> 1);
    dst[((nt * kTilesTotal + kt) << 9) + (lane << 4) + (j << 1) + (kk & 1)] = f2bfu(v);
}

// ---------------- persistent ODE-RNN kernel: 8 blocks x 512 threads (16 waves) ----------------
__global__ __launch_bounds__(512) void odernn_kernel(
    const float* __restrict__ ts, const float* __restrict__ X, const float* __restrict__ Stat,
    const unsigned short* __restrict__ wsb,
    const float* __restrict__ b0, const float* __restrict__ b1, const float* __restrict__ b2,
    const float* __restrict__ scale_p, const float* __restrict__ bx, const float* __restrict__ ib,
    const float* __restrict__ ob0, const float* __restrict__ ob1, const float* __restrict__ ob2,
    float* __restrict__ out) {

    __shared__ __align__(32) unsigned short panelA[8 * 512];   // 8 KB, up to K=256
    __shared__ __align__(32) unsigned short panelB[8 * 512];   // 8 KB
    __shared__ float ybuf[16 * 128];                           // 8 KB  (substep state y)
    __shared__ float hbuf[16 * 128];                           // 8 KB  (hidden state h)
    __shared__ unsigned short kb[6][16 * 128];                 // 24 KB (Tsit5 k1..k6, bf16)
    __shared__ float t0_s[16], dth_s[16];

    const int tid  = threadIdx.x;
    const int lane = tid & 31;
    const int wave = tid >> 5;
    const int m0   = blockIdx.x << 4;          // 16 batch rows per block
    const float scale = scale_p[0];

    const unsigned short* W0p  = wsb + OFF_W0;
    const unsigned short* W1p  = wsb + OFF_W1;
    const unsigned short* W2p  = wsb + OFF_W2;
    const unsigned short* RNNp = wsb + OFF_RNN;
    const unsigned short* INJp = wsb + OFF_INJ;
    const unsigned short* O0p  = wsb + OFF_O0;
    const unsigned short* O1p  = wsb + OFF_O1;
    const unsigned short* O2p  = wsb + OFF_O2;

    // ---- h0 = tanh(X[:,0] @ Wx^T + bx)  (use fused RNN weights; h-part of panel = 0) ----
    #pragma unroll 1
    for (int e = tid; e < 16 * 160; e += 512) {
        const int m = e / 160, k = e - m * 160;
        float v = 0.f;
        if (k >= 128) v = X[((m0 + m) * 64 + 0) * 32 + (k - 128)];
        panel_write(panelA, m, k, f2bfu(v));
    }
    __syncthreads();
    if (wave < 8) {
        const v8f acc = gemm16(panelA, RNNp + wave * (5 << 9), 5, lane);
        const int ncol = (wave << 4) + (lane & 15);
        const float bb = bx[ncol];
        #pragma unroll
        for (int r = 0; r < 8; ++r) {
            const int m = r + ((lane >> 4) << 3);
            hbuf[(m << 7) + ncol] = fast_tanh(acc[r] + bb);
        }
    }
    __syncthreads();

    // ---- h0 = tanh(concat(h0, Static) @ inj_w^T + inj_b) ----
    #pragma unroll 1
    for (int e = tid; e < 16 * 160; e += 512) {
        const int m = e / 160, k = e - m * 160;
        float v = 0.f;
        if (k < 128)       v = hbuf[(m << 7) + k];
        else if (k < 144)  v = Stat[(m0 + m) * 16 + (k - 128)];
        panel_write(panelA, m, k, f2bfu(v));
    }
    __syncthreads();
    if (wave < 8) {
        const v8f acc = gemm16(panelA, INJp + wave * (5 << 9), 5, lane);
        const int ncol = (wave << 4) + (lane & 15);
        const float bb = ib[ncol];
        #pragma unroll
        for (int r = 0; r < 8; ++r) {
            const int m = r + ((lane >> 4) << 3);
            hbuf[(m << 7) + ncol] = fast_tanh(acc[r] + bb);
        }
    }
    __syncthreads();

    // ---- recurrence over 63 intervals ----
    #pragma unroll 1
    for (int step = 0; step < 63; ++step) {
        if (tid < 16) {
            const float ta = ts[(m0 + tid) * 64 + step];
            const float tb = ts[(m0 + tid) * 64 + step + 1];
            t0_s[tid]  = ta;
            dth_s[tid] = (tb - ta) * 0.5f;     // dt per Tsit5 substep
        }
        #pragma unroll 1
        for (int e = tid; e < 2048; e += 512) ybuf[e] = hbuf[e];
        __syncthreads();

        #pragma unroll 1
        for (int sub = 0; sub < 2; ++sub) {
            #pragma unroll 1
            for (int st = 0; st < 6; ++st) {
                // build layer-1 input panel: z = [t_stage, y + d * sum(a_i * k_i)], Kpad=160
                const float cst = CTAB[st];
                #pragma unroll 1
                for (int e = tid; e < 16 * 160; e += 512) {
                    const int m = e / 160, k = e - m * 160;
                    float v;
                    if (k == 0) {
                        v = t0_s[m] + cst * dth_s[m];
                    } else if (k <= 128) {
                        const int j = k - 1;
                        float a = 0.f;
                        #pragma unroll 1
                        for (int i = 0; i < st; ++i)
                            a += ATAB[st][i] * bf2f(kb[i][(m << 7) + j]);
                        v = ybuf[(m << 7) + j] + dth_s[m] * a;
                    } else {
                        v = 0.f;
                    }
                    panel_write(panelA, m, k, f2bfu(v));
                }
                __syncthreads();

                // layer 1: (16x160)(160x256) + b0, silu -> panelB
                {
                    const v8f acc = gemm16(panelA, W0p + wave * (5 << 9), 5, lane);
                    const int ncol = (wave << 4) + (lane & 15);
                    const float bb = b0[ncol];
                    #pragma unroll
                    for (int r = 0; r < 8; ++r) {
                        const int m = r + ((lane >> 4) << 3);
                        panel_write(panelB, m, ncol, f2bfu(fast_silu(acc[r] + bb)));
                    }
                }
                __syncthreads();

                // layer 2: (16x256)(256x256) + b1, silu -> panelA
                {
                    const v8f acc = gemm16(panelB, W1p + wave * (8 << 9), 8, lane);
                    const int ncol = (wave << 4) + (lane & 15);
                    const float bb = b1[ncol];
                    #pragma unroll
                    for (int r = 0; r < 8; ++r) {
                        const int m = r + ((lane >> 4) << 3);
                        panel_write(panelA, m, ncol, f2bfu(fast_silu(acc[r] + bb)));
                    }
                }
                __syncthreads();

                // layer 3: (16x256)(256x128) + b2, scale*tanh -> kb[st]
                if (wave < 8) {
                    const v8f acc = gemm16(panelA, W2p + wave * (8 << 9), 8, lane);
                    const int ncol = (wave << 4) + (lane & 15);
                    const float bb = b2[ncol];
                    #pragma unroll
                    for (int r = 0; r < 8; ++r) {
                        const int m = r + ((lane >> 4) << 3);
                        kb[st][(m << 7) + ncol] = f2bfu(scale * fast_tanh(acc[r] + bb));
                    }
                }
                __syncthreads();
            }

            // y += d * sum(B_i * k_i); advance substep time
            #pragma unroll 1
            for (int e = tid; e < 2048; e += 512) {
                const int m = e >> 7;
                float a = 0.f;
                #pragma unroll
                for (int i = 0; i < 6; ++i) a += BTAB[i] * bf2f(kb[i][e]);
                ybuf[e] += dth_s[m] * a;
            }
            if (tid < 16) t0_s[tid] += dth_s[tid];
            __syncthreads();
        }

        // RNN cell: h = tanh(h' @ Wh^T + x @ Wx^T + bx) as one fused K=160 GEMM
        #pragma unroll 1
        for (int e = tid; e < 16 * 160; e += 512) {
            const int m = e / 160, k = e - m * 160;
            float v;
            if (k < 128) v = ybuf[(m << 7) + k];
            else         v = X[((m0 + m) * 64 + step + 1) * 32 + (k - 128)];
            panel_write(panelA, m, k, f2bfu(v));
        }
        __syncthreads();
        if (wave < 8) {
            const v8f acc = gemm16(panelA, RNNp + wave * (5 << 9), 5, lane);
            const int ncol = (wave << 4) + (lane & 15);
            const float bb = bx[ncol];
            #pragma unroll
            for (int r = 0; r < 8; ++r) {
                const int m = r + ((lane >> 4) << 3);
                hbuf[(m << 7) + ncol] = fast_tanh(acc[r] + bb);
            }
        }
        __syncthreads();
    }

    // ---- output head ----
    #pragma unroll 1
    for (int e = tid; e < 2048; e += 512) {
        const int m = e >> 7, k = e & 127;
        panel_write(panelA, m, k, f2bfu(hbuf[e]));
    }
    __syncthreads();
    {   // z = tanh(h @ out_w0^T + ob0): (16x128)(128x256) -> panelB
        const v8f acc = gemm16(panelA, O0p + wave * (4 << 9), 4, lane);
        const int ncol = (wave << 4) + (lane & 15);
        const float bb = ob0[ncol];
        #pragma unroll
        for (int r = 0; r < 8; ++r) {
            const int m = r + ((lane >> 4) << 3);
            panel_write(panelB, m, ncol, f2bfu(fast_tanh(acc[r] + bb)));
        }
    }
    __syncthreads();
    {   // z = tanh(z @ out_w1^T + ob1): (16x256)(256x256) -> panelA
        const v8f acc = gemm16(panelB, O1p + wave * (8 << 9), 8, lane);
        const int ncol = (wave << 4) + (lane & 15);
        const float bb = ob1[ncol];
        #pragma unroll
        for (int r = 0; r < 8; ++r) {
            const int m = r + ((lane >> 4) << 3);
            panel_write(panelA, m, ncol, f2bfu(fast_tanh(acc[r] + bb)));
        }
    }
    __syncthreads();
    if (wave == 0) {  // out = z @ out_w2^T + ob2: (16x256)(256x16) -> d_out (fp32)
        const v8f acc = gemm16(panelA, O2p, 8, lane);
        const int ncol = lane & 15;
        const float bb = ob2[ncol];
        #pragma unroll
        for (int r = 0; r < 8; ++r) {
            const int m = r + ((lane >> 4) << 3);
            out[(m0 + m) * 16 + ncol] = acc[r] + bb;
        }
    }
}

extern "C" void kernel_launch(void* const* d_in, const int* in_sizes, int n_in,
                              void* d_out, int out_size, void* d_ws, size_t ws_size,
                              hipStream_t stream) {
    (void)in_sizes; (void)n_in; (void)out_size; (void)ws_size;
    const float* ts   = (const float*)d_in[0];
    const float* X    = (const float*)d_in[1];
    const float* Stat = (const float*)d_in[2];
    const float* w0   = (const float*)d_in[3];
    const float* b0   = (const float*)d_in[4];
    const float* w1   = (const float*)d_in[5];
    const float* b1   = (const float*)d_in[6];
    const float* w2   = (const float*)d_in[7];
    const float* b2   = (const float*)d_in[8];
    const float* scal = (const float*)d_in[9];
    const float* wh   = (const float*)d_in[10];
    const float* wx   = (const float*)d_in[11];
    const float* bx   = (const float*)d_in[12];
    const float* iw   = (const float*)d_in[13];
    const float* ib   = (const float*)d_in[14];
    const float* ow0  = (const float*)d_in[15];
    const float* ob0  = (const float*)d_in[16];
    const float* ow1  = (const float*)d_in[17];
    const float* ob1  = (const float*)d_in[18];
    const float* ow2  = (const float*)d_in[19];
    const float* ob2  = (const float*)d_in[20];
    unsigned short* wsb = (unsigned short*)d_ws;

    auto pk = [&](const float* src, int off, int N, int Ksrc, int kOff, int kTiles, int Kspan) {
        const int total = N * Kspan;
        pack_weight<<<(total + 255) / 256, 256, 0, stream>>>(src, wsb + off, N, Ksrc, kOff,
                                                             kTiles, Kspan);
    };
    pk(w0,  OFF_W0,  256, 129, 0,   5, 160);
    pk(w1,  OFF_W1,  256, 256, 0,   8, 256);
    pk(w2,  OFF_W2,  128, 256, 0,   8, 256);
    pk(wh,  OFF_RNN, 128, 128, 0,   5, 128);   // Wh into K[0,128)
    pk(wx,  OFF_RNN, 128, 32,  128, 5, 32);    // Wx into K[128,160)
    pk(iw,  OFF_INJ, 128, 144, 0,   5, 160);
    pk(ow0, OFF_O0,  256, 128, 0,   4, 128);
    pk(ow1, OFF_O1,  256, 256, 0,   8, 256);
    pk(ow2, OFF_O2,  16,  256, 0,   8, 256);

    odernn_kernel<<<8, 512, 0, stream>>>(ts, X, Stat, wsb, b0, b1, b2, scal, bx, ib,
                                         ob0, ob1, ob2, (float*)d_out);
}
